// S2TATBlock_80951543595721
// MI455X (gfx1250) — compile-verified, matrix operand order checked by reading, not run on hardware
//
#include <hip/hip_runtime.h>
#include <hip/hip_bf16.h>

#define BB 8
#define TT 12
#define NN 1024
#define CC 64
#define HH 8
#define HD 512

typedef __attribute__((ext_vector_type(16))) __bf16 bf16x16;
typedef __attribute__((ext_vector_type(8)))  __bf16 bf16x8;
typedef __attribute__((ext_vector_type(8)))  float  f32x8;
typedef __attribute__((ext_vector_type(4)))  unsigned int u32x4;
typedef __attribute__((ext_vector_type(8)))  int i32x8;
typedef __attribute__((ext_vector_type(4)))  int i32x4;

#define LDSTR 40   // LDS row stride in bf16 elements (32 data + 8 pad) = 80B, keeps 16B alignment

// Build a 16x32 (or 32x16) bf16 WMMA fragment from LDS.
// Per CDNA5 ISA 16-bit A layout: lane = row (mod 16); lanes 0-15 hold K = {0..7,16..23},
// lanes 16-31 hold K = {8..15,24..31}. Two contiguous 16B chunks per lane.
__device__ __forceinline__ bf16x16 ld_frag(const __bf16* p, int hf) {
  bf16x8 lo = *(const bf16x8*)(p + 8 * hf);
  bf16x8 hi = *(const bf16x8*)(p + 16 + 8 * hf);
  bf16x16 r;
#pragma unroll
  for (int i = 0; i < 8; ++i) { r[i] = lo[i]; r[8 + i] = hi[i]; }
  return r;
}

// TDM: async-load a 128x32 bf16 tile (row stride 1024 elements) into LDS with
// 16B padding after each 64B row -> matches the As[128][LDSTR=40] layout.
// D# group0: count=1, lds_addr, 57b global_addr, type=2.
// D# group1: data_size=1(2B), pad_enable, pad_interval=3 (16 DW), pad_amount=3 (4 DW),
//            tensor_dim0=1024, tensor_dim1=1024, tile_dim0=32, tile_dim1=128,
//            tensor_dim0_stride=1024.
__device__ __forceinline__ void tdm_load_tile_128x32_bf16(const __bf16* gsrc, unsigned lds_addr) {
  unsigned long long ga = (unsigned long long)(uintptr_t)gsrc;
  u32x4 g0;
  g0[0] = 1u;                                   // count=1 (valid), user mode
  g0[1] = lds_addr;                             // LDS byte address
  g0[2] = (unsigned)(ga & 0xFFFFFFFFu);         // global_addr[31:0]
  g0[3] = (unsigned)(ga >> 32) | (2u << 30);    // global_addr[56:32] | type=2
  i32x8 g1;
  g1[0] = (1 << 16) | (1 << 20) | (3 << 22) | (3 << 25); // data_size=2B, pad en, intv=16DW, amt=4DW
  g1[1] = (int)(1024u << 16);                   // tensor_dim0 = 1024 (low 16 in [31:16])
  g1[2] = (int)(1024u << 16);                   // tensor_dim0 hi=0 | tensor_dim1 lo = 1024
  g1[3] = (int)(32u << 16);                     // tensor_dim1 hi=0 | tile_dim0 = 32
  g1[4] = 128;                                  // tile_dim1 = 128, tile_dim2 = 0
  g1[5] = 1024;                                 // tensor_dim0_stride[31:0] = 1024
  g1[6] = 0;                                    // stride hi | tensor_dim1_stride lo
  g1[7] = 0;
  i32x4 z4 = {0, 0, 0, 0};
#if __clang_major__ >= 23
  i32x8 z8 = {0, 0, 0, 0, 0, 0, 0, 0};
  __builtin_amdgcn_tensor_load_to_lds(g0, g1, z4, z4, z8, 0);
#else
  __builtin_amdgcn_tensor_load_to_lds(g0, g1, z4, z4, 0);
#endif
}

// ---------------------------------------------------------------- utilities
__global__ void cvt_bf16_kernel(const float* __restrict__ src, __bf16* __restrict__ dst, int n) {
  int i = blockIdx.x * blockDim.x + threadIdx.x;
  if (i < n) dst[i] = (__bf16)src[i];
}

// ---------------------------------------------------------------- q/k projections
__global__ __launch_bounds__(256) void qk_proj_kernel(const float* __restrict__ gs,
                                                      const float* __restrict__ Wq,
                                                      const float* __restrict__ bq,
                                                      const float* __restrict__ Wk,
                                                      const float* __restrict__ bk,
                                                      float* __restrict__ q,
                                                      float* __restrict__ k) {
  int bt = blockIdx.x;
  __shared__ float xs[NN];
  for (int i = threadIdx.x; i < NN; i += 256) xs[i] = gs[bt * NN + i];
  __syncthreads();
  for (int j = threadIdx.x; j < HD; j += 256) {
    float aq = 0.f, ak = 0.f;
    for (int m = 0; m < NN; ++m) {
      float v = xs[m];
      aq += v * Wq[m * HD + j];
      ak += v * Wk[m * HD + j];
    }
    q[bt * HD + j] = aq + bq[j];
    k[bt * HD + j] = ak + bk[j];
  }
}

// ---------------------------------------------------------------- attention weights (12x12 per (h,b))
__global__ __launch_bounds__(256) void attn_weights_kernel(const float* __restrict__ q,
                                                           const float* __restrict__ k,
                                                           float* __restrict__ w) {
  int hb = blockIdx.x;          // h*8 + b
  int h = hb >> 3, b = hb & 7;
  __shared__ float qs[TT * 64], ks[TT * 64], sc[TT * TT];
  for (int i = threadIdx.x; i < TT * 64; i += 256) {
    int t = i >> 6, d = i & 63;
    qs[i] = q[(b * TT + t) * HD + h * 64 + d];
    ks[i] = k[(b * TT + t) * HD + h * 64 + d];
  }
  __syncthreads();
  if (threadIdx.x < TT * TT) {
    int t = threadIdx.x / TT, tp = threadIdx.x % TT;
    float s = 0.f;
    for (int d = 0; d < 64; ++d) s += qs[t * 64 + d] * ks[tp * 64 + d];
    sc[threadIdx.x] = s * 0.125f;   // 1/sqrt(64)
  }
  __syncthreads();
  if (threadIdx.x < TT) {
    int t = threadIdx.x;
    float m = -1e30f;
    for (int tp = 0; tp < TT; ++tp) m = fmaxf(m, sc[t * TT + tp]);
    float e[TT], sum = 0.f;
    for (int tp = 0; tp < TT; ++tp) { e[tp] = __expf(sc[t * TT + tp] - m); sum += e[tp]; }
    float inv = 1.f / sum;
    for (int tp = 0; tp < TT; ++tp) w[(hb * TT + t) * TT + tp] = e[tp] * inv;
  }
}

// ---------------------------------------------------------------- WMMA GEMM: v1T[bt,hd,n] = (signal[bt] @ Kg[t])^T
// M=1024(n), K=64(c), N=512(hd).  grid = 96*32, block tile 128x128.
// Output written TRANSPOSED so adj_v's B tile becomes contiguous vector loads.
__global__ __launch_bounds__(256) void gcn_v_kernel(const float* __restrict__ gsig,
                                                    const float* __restrict__ Kg,
                                                    __bf16* __restrict__ v1T) {
  __shared__ __bf16 As[128 * LDSTR];
  __shared__ __bf16 Bs[128 * LDSTR];   // transposed: [hd][c]
  int bt = blockIdx.x >> 5;
  int tile = blockIdx.x & 31;
  int mt = tile >> 2, nt = tile & 3;
  int t = bt % TT;
  const float* Ag = gsig + (size_t)bt * NN * CC + (size_t)mt * 128 * CC;
  const float* Bg = Kg + (size_t)t * CC * HD + nt * 128;
  int tid = threadIdx.x, lane = tid & 31, wave = tid >> 5;
  int lr = lane & 15, hf = lane >> 4;
  int wm = wave >> 2, wn = wave & 3;       // 2x4 wave grid -> 64x32 per wave
  f32x8 zero = {0.f, 0.f, 0.f, 0.f, 0.f, 0.f, 0.f, 0.f};
  f32x8 acc[4][2];
#pragma unroll
  for (int i = 0; i < 4; ++i)
#pragma unroll
    for (int j = 0; j < 2; ++j) acc[i][j] = zero;

  for (int kk = 0; kk < CC; kk += 32) {
    { // A: 128x32 f32 -> bf16
      int row = tid >> 1, kc = (tid & 1) * 16;
      const float* s = Ag + row * CC + kk + kc;
      __bf16* d = As + row * LDSTR + kc;
#pragma unroll
      for (int i = 0; i < 16; ++i) d[i] = (__bf16)s[i];
    }
    { // B: 32x128 f32 -> bf16, transposed into [hd][c]
      int k = tid >> 3, nc = (tid & 7) * 16;
      const float* s = Bg + (kk + k) * HD + nc;
#pragma unroll
      for (int i = 0; i < 16; ++i) Bs[(nc + i) * LDSTR + k] = (__bf16)s[i];
    }
    __syncthreads();
    bf16x16 bfr[2];
#pragma unroll
    for (int j = 0; j < 2; ++j) bfr[j] = ld_frag(Bs + (wn * 32 + j * 16 + lr) * LDSTR, hf);
#pragma unroll
    for (int i = 0; i < 4; ++i) {
      bf16x16 af = ld_frag(As + (wm * 64 + i * 16 + lr) * LDSTR, hf);
#pragma unroll
      for (int j = 0; j < 2; ++j)
        acc[i][j] = __builtin_amdgcn_wmma_f32_16x16x32_bf16(false, af, false, bfr[j],
                                                            (short)0, acc[i][j], false, false);
    }
    __syncthreads();
  }
  // Transposed store: per lane, fixed column (hd), 8 consecutive rows (n) -> one 16B store.
#pragma unroll
  for (int i = 0; i < 4; ++i)
#pragma unroll
    for (int j = 0; j < 2; ++j) {
      bf16x8 pkt;
#pragma unroll
      for (int r = 0; r < 8; ++r) pkt[r] = (__bf16)acc[i][j][r];
      int col = nt * 128 + wn * 32 + j * 16 + lr;             // hd
      int row0 = mt * 128 + wm * 64 + i * 16 + hf * 8;        // n
      *(bf16x8*)(v1T + ((size_t)bt * HD + col) * NN + row0) = pkt;
    }
}

// ---------------------------------------------------------------- WMMA GEMM: v2[bt] = adj @ v1[bt] + bg[t]
// M=1024, K=1024, N=512.  Dominant 103 GFLOP stage.
// A tile loaded by the Tensor Data Mover (async DMA -> LDS, padded rows);
// B tile vector-copied from the transposed v1T.
__global__ __launch_bounds__(256) void adj_v_kernel(const __bf16* __restrict__ adjb,
                                                    const __bf16* __restrict__ v1T,
                                                    const float* __restrict__ bg,
                                                    __bf16* __restrict__ v2) {
  __shared__ __bf16 As[128 * LDSTR];
  __shared__ __bf16 Bs[128 * LDSTR];
  int bt = blockIdx.x >> 5;
  int tile = blockIdx.x & 31;
  int mt = tile >> 2, nt = tile & 3;
  int t = bt % TT;
  const __bf16* Ag = adjb + (size_t)mt * 128 * NN;
  const __bf16* BgT = v1T + ((size_t)bt * HD + (size_t)nt * 128) * NN;
  int tid = threadIdx.x, lane = tid & 31, wave = tid >> 5;
  int lr = lane & 15, hf = lane >> 4;
  int wm = wave >> 2, wn = wave & 3;
  unsigned as_lds = (unsigned)(uintptr_t)(&As[0]);
  f32x8 zero = {0.f, 0.f, 0.f, 0.f, 0.f, 0.f, 0.f, 0.f};
  f32x8 acc[4][2];
#pragma unroll
  for (int i = 0; i < 4; ++i)
#pragma unroll
    for (int j = 0; j < 2; ++j) acc[i][j] = zero;

  for (int kk = 0; kk < NN; kk += 32) {
    if (wave == 0) {  // one TDM descriptor per block per k-step; TDM ignores EXEC
      tdm_load_tile_128x32_bf16(Ag + kk, as_lds);
      __builtin_amdgcn_s_wait_tensorcnt(0);
    }
    { // B: 128(hd) x 32(k) vector copy from v1T (contiguous in k)
      int row = tid >> 1, kc = (tid & 1) * 16;
      const __bf16* s = BgT + (size_t)row * NN + kk + kc;
      uint4* d = (uint4*)(Bs + row * LDSTR + kc);
      d[0] = ((const uint4*)s)[0];
      d[1] = ((const uint4*)s)[1];
      if (kk + 32 < NN) __builtin_prefetch(s + 32, 0, 3);   // next k-step -> global_prefetch_b8
    }
    __syncthreads();
    bf16x16 bfr[2];
#pragma unroll
    for (int j = 0; j < 2; ++j) bfr[j] = ld_frag(Bs + (wn * 32 + j * 16 + lr) * LDSTR, hf);
#pragma unroll
    for (int i = 0; i < 4; ++i) {
      bf16x16 af = ld_frag(As + (wm * 64 + i * 16 + lr) * LDSTR, hf);
#pragma unroll
      for (int j = 0; j < 2; ++j)
        acc[i][j] = __builtin_amdgcn_wmma_f32_16x16x32_bf16(false, af, false, bfr[j],
                                                            (short)0, acc[i][j], false, false);
    }
    __syncthreads();
  }
#pragma unroll
  for (int i = 0; i < 4; ++i)
#pragma unroll
    for (int j = 0; j < 2; ++j)
#pragma unroll
      for (int r = 0; r < 8; ++r) {
        int row = mt * 128 + wm * 64 + i * 16 + hf * 8 + r;
        int col = nt * 128 + wn * 32 + j * 16 + lr;
        float val = acc[i][j][r] + bg[t * HD + col];
        v2[((size_t)bt * NN + row) * HD + col] = (__bf16)val;
      }
}

// ---------------------------------------------------------------- T-mixing: update[b,t,n,hd] = sum_tp w[h,b,t,tp]*v2[b,tp,n,hd]
__global__ __launch_bounds__(256) void attn_mix_kernel(const __bf16* __restrict__ v2,
                                                       const float* __restrict__ w,
                                                       __bf16* __restrict__ upd) {
  int bn = blockIdx.x;            // b*1024 + n
  int b = bn >> 10, n = bn & 1023;
  __shared__ float wl[HH * TT * TT];
  for (int i = threadIdx.x; i < HH * TT * TT; i += 256) {
    int h = i / (TT * TT);
    wl[i] = w[(h * BB + b) * TT * TT + (i % (TT * TT))];
  }
  __syncthreads();
  for (int hd = threadIdx.x; hd < HD; hd += 256) {
    int h = hd >> 6;
    float acc[TT];
#pragma unroll
    for (int t = 0; t < TT; ++t) acc[t] = 0.f;
    for (int tp = 0; tp < TT; ++tp) {
      float vv = (float)v2[(((size_t)b * TT + tp) * NN + n) * HD + hd];
#pragma unroll
      for (int t = 0; t < TT; ++t) acc[t] += wl[h * TT * TT + t * TT + tp] * vv;
    }
#pragma unroll
    for (int t = 0; t < TT; ++t)
      upd[(((size_t)b * TT + t) * NN + n) * HD + hd] = (__bf16)acc[t];
  }
}

// ---------------------------------------------------------------- state projection (HD -> 1) + relu + residual
__global__ __launch_bounds__(256) void state_proj_kernel(const __bf16* __restrict__ upd,
                                                         const float* __restrict__ Wsp,
                                                         const float* __restrict__ bsp,
                                                         const float* __restrict__ gstate,
                                                         float* __restrict__ out_state) {
  __shared__ float wsl[HD];
  for (int i = threadIdx.x; i < HD; i += 256) wsl[i] = Wsp[i];
  __syncthreads();
  int wave = threadIdx.x >> 5, lane = threadIdx.x & 31;
  int row = blockIdx.x * 8 + wave;               // 12288 blocks * 8 waves = 98304 rows
  const __bf16* up = upd + (size_t)row * HD + lane * 16;
  bf16x8 u0 = *(const bf16x8*)(up);
  bf16x8 u1 = *(const bf16x8*)(up + 8);
  float s = 0.f;
#pragma unroll
  for (int i = 0; i < 8; ++i) {
    s += (float)u0[i] * wsl[lane * 16 + i];
    s += (float)u1[i] * wsl[lane * 16 + 8 + i];
  }
  for (int off = 16; off > 0; off >>= 1) s += __shfl_xor(s, off, 32);
  if (lane == 0) out_state[row] = gstate[row] + fmaxf(s + bsp[0], 0.f);
}

// ---------------------------------------------------------------- WMMA GEMM: signal_ = relu(update @ Wf + bf) + residual
// M=98304, K=512, N=64.  Block tile 128x64, 8 waves in 4x2, writes fp32 pre-LN into d_out.
__global__ __launch_bounds__(256) void signal_proj_kernel(const __bf16* __restrict__ upd,
                                                          const float* __restrict__ Wf,
                                                          const float* __restrict__ bfv,
                                                          const float* __restrict__ gsig,
                                                          float* __restrict__ out_sig) {
  __shared__ __bf16 As[128 * LDSTR];
  __shared__ __bf16 Bs[64 * LDSTR];
  int row0 = blockIdx.x * 128;
  int tid = threadIdx.x, lane = tid & 31, wave = tid >> 5;
  int lr = lane & 15, hf = lane >> 4;
  int wm = wave >> 1, wn = wave & 1;   // 4x2 wave grid -> 32x32 per wave
  f32x8 zero = {0.f, 0.f, 0.f, 0.f, 0.f, 0.f, 0.f, 0.f};
  f32x8 acc[2][2];
#pragma unroll
  for (int i = 0; i < 2; ++i)
#pragma unroll
    for (int j = 0; j < 2; ++j) acc[i][j] = zero;

  for (int kk = 0; kk < HD; kk += 32) {
    { // A: 128x32 bf16 from update
      int row = tid >> 1, kc = (tid & 1) * 16;
      const uint4* s = (const uint4*)(upd + (size_t)(row0 + row) * HD + kk + kc);
      uint4* d = (uint4*)(As + row * LDSTR + kc);
      d[0] = s[0]; d[1] = s[1];
    }
    { // B: 32x64 f32 Wf -> transposed bf16 [c][k]
      int k = tid >> 3, nc = (tid & 7) * 8;
      const float* s = Wf + (kk + k) * CC + nc;
#pragma unroll
      for (int i = 0; i < 8; ++i) Bs[(nc + i) * LDSTR + k] = (__bf16)s[i];
    }
    __syncthreads();
    bf16x16 bfr[2];
#pragma unroll
    for (int j = 0; j < 2; ++j) bfr[j] = ld_frag(Bs + (wn * 32 + j * 16 + lr) * LDSTR, hf);
#pragma unroll
    for (int i = 0; i < 2; ++i) {
      bf16x16 af = ld_frag(As + (wm * 32 + i * 16 + lr) * LDSTR, hf);
#pragma unroll
      for (int j = 0; j < 2; ++j)
        acc[i][j] = __builtin_amdgcn_wmma_f32_16x16x32_bf16(false, af, false, bfr[j],
                                                            (short)0, acc[i][j], false, false);
    }
    __syncthreads();
  }
#pragma unroll
  for (int i = 0; i < 2; ++i)
#pragma unroll
    for (int j = 0; j < 2; ++j)
#pragma unroll
      for (int r = 0; r < 8; ++r) {
        int row = row0 + wm * 32 + i * 16 + hf * 8 + r;
        int col = wn * 32 + j * 16 + lr;
        float v = fmaxf(acc[i][j][r] + bfv[col], 0.f);
        out_sig[(size_t)row * CC + col] = gsig[(size_t)row * CC + col] + v;
      }
}

// ---------------------------------------------------------------- deterministic two-stage LayerNorm
__global__ __launch_bounds__(256) void partial_moments_kernel(const float* __restrict__ x,
                                                              int per_b, int blocks_per_b,
                                                              float* __restrict__ partials) {
  int b = blockIdx.x / blocks_per_b;
  int chunk = blockIdx.x % blocks_per_b;
  int len = per_b / blocks_per_b;
  const float* p = x + (size_t)b * per_b + (size_t)chunk * len;
  float s = 0.f, ss = 0.f;
  for (int i = threadIdx.x; i < len; i += 256) { float v = p[i]; s += v; ss += v * v; }
  __shared__ float sh[256], sh2[256];
  sh[threadIdx.x] = s; sh2[threadIdx.x] = ss;
  __syncthreads();
  for (int o = 128; o > 0; o >>= 1) {
    if (threadIdx.x < o) { sh[threadIdx.x] += sh[threadIdx.x + o]; sh2[threadIdx.x] += sh2[threadIdx.x + o]; }
    __syncthreads();
  }
  if (threadIdx.x == 0) {
    partials[2 * blockIdx.x] = sh[0];
    partials[2 * blockIdx.x + 1] = sh2[0];
  }
}

__global__ void finalize_moments_kernel(const float* __restrict__ partials, int blocks_per_b,
                                        float inv_n, float* __restrict__ stats) {
  int b = blockIdx.x;
  if (threadIdx.x == 0) {
    float s = 0.f, ss = 0.f;
    for (int i = 0; i < blocks_per_b; ++i) {
      s += partials[2 * (b * blocks_per_b + i)];
      ss += partials[2 * (b * blocks_per_b + i) + 1];
    }
    float m = s * inv_n;
    float var = ss * inv_n - m * m;
    stats[2 * b] = m;
    stats[2 * b + 1] = rsqrtf(var + 1e-3f);
  }
}

__global__ __launch_bounds__(256) void norm_kernel(float* __restrict__ x,
                                                   const float* __restrict__ stats,
                                                   const float* __restrict__ gamma,
                                                   const float* __restrict__ beta,
                                                   int per_b) {
  size_t idx = (size_t)blockIdx.x * 256 + threadIdx.x;
  int b = (int)(idx / per_b);
  int r = (int)(idx % per_b);
  float m = stats[2 * b], rs = stats[2 * b + 1];
  x[idx] = (x[idx] - m) * rs * gamma[r] + beta[r];
}

// ---------------------------------------------------------------- launch
extern "C" void kernel_launch(void* const* d_in, const int* in_sizes, int n_in,
                              void* d_out, int out_size, void* d_ws, size_t ws_size,
                              hipStream_t stream) {
  (void)in_sizes; (void)n_in; (void)out_size; (void)ws_size;
  const float* gstate = (const float*)d_in[0];
  const float* gsig   = (const float*)d_in[1];
  const float* adj    = (const float*)d_in[2];
  const float* Wq     = (const float*)d_in[3];
  const float* bq     = (const float*)d_in[4];
  const float* Wk     = (const float*)d_in[5];
  const float* bk     = (const float*)d_in[6];
  const float* Kg     = (const float*)d_in[7];
  const float* bg     = (const float*)d_in[8];
  const float* Wsp    = (const float*)d_in[9];
  const float* bsp    = (const float*)d_in[10];
  const float* Wf     = (const float*)d_in[11];
  const float* bfv    = (const float*)d_in[12];
  const float* gs_g   = (const float*)d_in[13];
  const float* gs_b   = (const float*)d_in[14];
  const float* sg_g   = (const float*)d_in[15];
  const float* sg_b   = (const float*)d_in[16];

  char* ws = (char*)d_ws;
  float*  q      = (float*)(ws + 0);                       // 96*512 f32
  float*  k      = (float*)(ws + (256u << 10));            // 96*512 f32
  float*  w      = (float*)(ws + (512u << 10));            // 64*144 f32
  float*  part_s = (float*)(ws + (576u << 10));            // 8*8*2 f32
  float*  part_g = (float*)(ws + (576u << 10) + 4096);     // 8*64*2 f32
  float*  stat_s = (float*)(ws + (576u << 10) + 16384);    // 16 f32
  float*  stat_g = (float*)(ws + (576u << 10) + 16640);    // 16 f32
  __bf16* adjb   = (__bf16*)(ws + (1u << 20));             // 1024*1024 bf16 (2 MB)
  __bf16* v1T    = (__bf16*)(ws + (4u << 20));             // 96*512*1024 bf16 (96 MB), [bt][hd][n]
  __bf16* v2     = (__bf16*)(ws + (4u << 20) + (size_t)100663296); // 96 MB, [bt][n][hd]
  __bf16* upd    = v1T;                                    // v1T dead after adj_v -> reuse memory

  float* out_state = (float*)d_out;                        // [B,T,N]
  float* out_sig   = (float*)d_out + BB * TT * NN;         // [B,T,N,C]

  cvt_bf16_kernel<<<4096, 256, 0, stream>>>(adj, adjb, NN * NN);
  qk_proj_kernel<<<BB * TT, 256, 0, stream>>>(gstate, Wq, bq, Wk, bk, q, k);
  attn_weights_kernel<<<HH * BB, 256, 0, stream>>>(q, k, w);
  gcn_v_kernel<<<BB * TT * 32, 256, 0, stream>>>(gsig, Kg, v1T);
  adj_v_kernel<<<BB * TT * 32, 256, 0, stream>>>(adjb, v1T, bg, v2);
  attn_mix_kernel<<<BB * NN, 256, 0, stream>>>(v2, w, upd);
  state_proj_kernel<<<BB * TT * NN / 8, 256, 0, stream>>>(upd, Wsp, bsp, gstate, out_state);
  signal_proj_kernel<<<BB * TT * NN / 128, 256, 0, stream>>>(upd, Wf, bfv, gsig, out_sig);

  partial_moments_kernel<<<BB * 8, 256, 0, stream>>>(out_state, TT * NN, 8, part_s);
  finalize_moments_kernel<<<BB, 32, 0, stream>>>(part_s, 8, 1.f / (TT * NN), stat_s);
  partial_moments_kernel<<<BB * 64, 256, 0, stream>>>(out_sig, TT * NN * CC, 64, part_g);
  finalize_moments_kernel<<<BB, 32, 0, stream>>>(part_g, 64, 1.f / (TT * NN * CC), stat_g);
  norm_kernel<<<BB * TT * NN / 256, 256, 0, stream>>>(out_state, stat_s, gs_g, gs_b, TT * NN);
  norm_kernel<<<BB * TT * NN * CC / 256, 256, 0, stream>>>(out_sig, stat_g, sg_g, sg_b, TT * NN * CC);
}
